// AF_LSTM_68934225100809
// MI455X (gfx1250) — matrix operand order repaired and checked by reading, not compiled, not run on hardware
//
#include <hip/hip_runtime.h>
#include <cstddef>
#include <cstdint>

// ---------------------------------------------------------------------------
// Problem constants (V=50000, D=256, T=512, B=64)
// ---------------------------------------------------------------------------
constexpr int kB = 64;
constexpr int kT = 512;
constexpr int kD = 256;
constexpr int kG = 1024;  // 4*D

typedef __bf16 bf16_t;
typedef __attribute__((ext_vector_type(16))) __bf16 bf16x16;
typedef __attribute__((ext_vector_type(8)))  __bf16 bf16x8;
typedef __attribute__((ext_vector_type(8)))  float  f32x8;

union FA { bf16x16 v; bf16_t e[16]; };
union FC { f32x8   v; float  e[8];  };

// ---------------------------------------------------------------------------
// Workspace layout (bytes).
// ---------------------------------------------------------------------------
constexpr size_t OFF_WIH = 0;                                   // [1024*256] bf16 (== Wt[n][k])
constexpr size_t OFF_WHH = OFF_WIH + (size_t)kG * kD * 2;       // [1024*256] bf16
constexpr size_t OFF_WYT = OFF_WHH + (size_t)kG * kD * 2;       // [256*256]  bf16 (W_y transposed)
constexpr size_t OFF_WP  = OFF_WYT + (size_t)kD * kD * 2;       // [256*256]  bf16
constexpr size_t OFF_WX  = OFF_WP  + (size_t)kD * kD * 2;
constexpr size_t OFF_WF  = OFF_WX  + (size_t)kD * kD * 2;
constexpr size_t OFF_XS  = OFF_WF  + (size_t)kD * kD * 2;       // [B*T,4D] bf16
constexpr size_t OFF_H   = OFF_XS  + (size_t)kB * kT * kG * 2;  // [B*T,D]  f32
constexpr size_t OFF_SN  = OFF_H   + (size_t)kB * kT * kD * 4;  // [B,D]    f32
constexpr size_t OFF_E   = OFF_SN  + (size_t)kB * kD * 4;       // [B,T]    f32
constexpr size_t OFF_R   = OFF_E   + (size_t)kB * kT * 4;       // [B,D]    f32

// ---------------------------------------------------------------------------
// WMMA helpers (CDNA5 wave32, V_WMMA_F32_16X16X32_BF16)
// ---------------------------------------------------------------------------
__device__ __forceinline__ f32x8 wmma_bf16(bf16x16 a, bf16x16 b, f32x8 c) {
  // 8 args: (neg_a, A, neg_b, B, c_mod, C, reuse_a, reuse_b)
  return __builtin_amdgcn_wmma_f32_16x16x32_bf16(false, a, false, b, (short)0, c,
                                                 false, false);
}

// Async global->LDS DMA (GLOBAL_LOAD_ASYNC_TO_LDS_B128, tracked by ASYNCcnt).
// lds_off = byte offset of destination inside the LDS aperture (low 32 bits of
// the generic shared address), gptr = per-lane global source address.
__device__ __forceinline__ void async_copy_b128(uint32_t lds_off, const void* gptr) {
  asm volatile("global_load_async_to_lds_b128 %0, %1, off"
               :: "v"(lds_off), "v"((uint64_t)(uintptr_t)gptr)
               : "memory");
}
__device__ __forceinline__ void wait_async0() {
  asm volatile("s_wait_asynccnt 0" ::: "memory");
}

// A fragment: 16x32 bf16 tile, row-major with leading dim lda (elements).
// Lanes 0-15 row M=lane hold K=0..15; lanes 16-31 hold K=16..31.
__device__ __forceinline__ bf16x16 frag_a_row(const bf16_t* A, int lda, int lane) {
  const int m = lane & 15, hs = lane >> 4;
  const bf16_t* p = A + m * lda + hs * 16;
  FA u;
  *(bf16x8*)&u.e[0] = *(const bf16x8*)(p);
  *(bf16x8*)&u.e[8] = *(const bf16x8*)(p + 8);
  return u.v;
}

// A fragment gathered from f32 memory with on-the-fly bf16 conversion.
__device__ __forceinline__ bf16x16 frag_a_f32(const float* A, int lda, int lane) {
  const int m = lane & 15, hs = lane >> 4;
  FA u;
#pragma unroll
  for (int i = 0; i < 16; ++i) {
    const int k = (hs << 4) + i;
    u.e[i] = (bf16_t)A[m * lda + k];
  }
  return u.v;
}

// B fragment: K=32 x N=16 from pre-transposed weights Wt[n][k] (ldk elements).
__device__ __forceinline__ bf16x16 frag_b_tr(const bf16_t* Wt, int ldk, int lane) {
  const int n = lane & 15, k0 = (lane >> 4) << 4;
  const bf16_t* p = Wt + n * ldk + k0;
  FA u;
  *(bf16x8*)&u.e[0] = *(const bf16x8*)(p);
  *(bf16x8*)&u.e[8] = *(const bf16x8*)(p + 8);
  return u.v;
}

// B fragment synthesized from the circulant of s_norm: B[k][n] = sn[(n+k)&255].
__device__ __forceinline__ bf16x16 frag_b_circ(const bf16_t* sn, int n0, int kbase,
                                               int lane) {
  const int n = n0 + (lane & 15);
  const int k0 = kbase + ((lane >> 4) << 4);
  FA u;
#pragma unroll
  for (int i = 0; i < 16; ++i) u.e[i] = sn[(n + k0 + i) & 255];
  return u.v;
}

__device__ __forceinline__ float sigm(float x) { return 1.0f / (1.0f + __expf(-x)); }

// ---------------------------------------------------------------------------
// Weight conversion kernels
// ---------------------------------------------------------------------------
__global__ void k_cvt(const float* __restrict__ src, bf16_t* __restrict__ dst, int n) {
  const int i = blockIdx.x * blockDim.x + threadIdx.x;
  if (i < n) dst[i] = (bf16_t)src[i];
}

__global__ void k_cvt_t(const float* __restrict__ src, bf16_t* __restrict__ dst) {
  // src [256][256] row-major; dst[n][k] = src[k][n]
  const int i = blockIdx.x * blockDim.x + threadIdx.x;
  const int k = i >> 8, n = i & 255;
  dst[n * kD + k] = (bf16_t)src[i];
}

// ---------------------------------------------------------------------------
// xs = emb[x] @ w_ih^T + b_lstm     [B*T, 4D]  (bf16 out, f32 accumulate)
// grid (2048 m-tiles, 8 n-groups), 256 threads = 8 waves (1 tile per wave)
// ---------------------------------------------------------------------------
__global__ void __launch_bounds__(256) k_xproj(const int* __restrict__ x,
                                               const float* __restrict__ emb,
                                               const bf16_t* __restrict__ wih,
                                               const float* __restrict__ bl,
                                               bf16_t* __restrict__ xs) {
  __shared__ __align__(16) bf16_t As[16 * 264];
  const int tid = threadIdx.x, lane = tid & 31, wave = tid >> 5;
  const int mt = blockIdx.x;  // 16 rows of flattened [B*T]
  {   // gather 16 embedding rows, convert to bf16 in LDS
    const int m = tid >> 4;
    const int d0 = (tid & 15) * 16;
    const int tok = x[mt * 16 + m];
    const float* src = emb + (size_t)tok * kD + d0;
    bf16_t* dst = As + m * 264 + d0;
#pragma unroll
    for (int j = 0; j < 16; ++j) dst[j] = (bf16_t)src[j];
  }
  __syncthreads();

  const int n0 = (blockIdx.y * 8 + wave) * 16;
  const int hs = lane >> 4, nl = lane & 15;
  FC acc;
  const float bias = bl[n0 + nl];
#pragma unroll
  for (int i = 0; i < 8; ++i) acc.e[i] = bias;
#pragma unroll
  for (int kk = 0; kk < 8; ++kk) {
    bf16x16 a = frag_a_row(As + kk * 32, 264, lane);
    bf16x16 b = frag_b_tr(wih + (size_t)n0 * kD + kk * 32, kD, lane);
    acc.v = wmma_bf16(a, b, acc.v);
  }
#pragma unroll
  for (int i = 0; i < 8; ++i) {
    const int m = i + 8 * hs;
    xs[(size_t)(mt * 16 + m) * kG + n0 + nl] = (bf16_t)acc.e[i];
  }
}

// ---------------------------------------------------------------------------
// Persistent LSTM scan. grid=4 (16 batches each), 512 threads = 16 waves.
// State in LDS for all 512 steps: h as bf16 A-fragments, c as f32.
// Per step:
//   0) xs_t tile (16x1024 bf16, 32KB) DMA'd into Gs via
//      GLOBAL_LOAD_ASYNC_TO_LDS_B128 (coalesced, no VGPR round-trip),
//      s_wait_asynccnt 0 + barrier.
//   1) A-fragments of h hoisted into VGPRs once, reused across 4 n-tiles:
//      G = xs_t + h @ w_hh^T (64 WMMA n-tiles, 8 K-steps each).
//   2) fused gate nonlinearities update (h,c), stream h to HBM.
// w_hh (512 KB bf16) stays L2-resident and is prefetched each step.
// ---------------------------------------------------------------------------
__global__ void __launch_bounds__(512) k_lstm(const bf16_t* __restrict__ xs,
                                              const bf16_t* __restrict__ whh,
                                              float* __restrict__ h_out) {
  __shared__ __align__(16) bf16_t Hs[16 * 264];   // hidden state (bf16)
  __shared__ __align__(16) bf16_t Gs[16 * 1024];  // xs_t staging + gate pre-acts
  __shared__ float Cs[16 * 256];                  // cell state (f32)
  const int tid = threadIdx.x, lane = tid & 31, wave = tid >> 5;
  const int b0 = blockIdx.x * 16;
  for (int i = tid; i < 16 * 264; i += 512) Hs[i] = (bf16_t)0.0f;
  for (int i = tid; i < 16 * 256; i += 512) Cs[i] = 0.0f;
  __syncthreads();

  const int hs_ = lane >> 4, nl = lane & 15;
  // async-copy geometry: 16 rows x 2KB; 32 threads/row x 64B each (4x b128)
  const int m_cp = tid >> 5;
  const int c_cp = tid & 31;
  const uint32_t gs_lds = (uint32_t)(uintptr_t)(&Gs[0]) +
                          (uint32_t)(m_cp * 2048 + c_cp * 64);
  const bf16_t* xs_row = xs + (size_t)(b0 + m_cp) * kT * kG + c_cp * 32;

  for (int t = 0; t < kT; ++t) {
    // ---- phase 0: DMA xs[:,t,:] -> Gs (ASYNCcnt path) ----
    {
      const char* src = (const char*)(xs_row + (size_t)t * kG);
#pragma unroll
      for (int j = 0; j < 4; ++j) async_copy_b128(gs_lds + j * 16, src + j * 16);
      wait_async0();
    }
    __syncthreads();

    // hoist the 8 A-fragments of h for this step (reused by all 4 n-tiles)
    bf16x16 afr[8];
#pragma unroll
    for (int kk = 0; kk < 8; ++kk) afr[kk] = frag_a_row(Hs + kk * 32, 264, lane);

    // ---- phase 1: G = xs_t + Hs @ whh^T ----
#pragma unroll
    for (int j = 0; j < 4; ++j) {
      const int n0 = (wave + 16 * j) * 16;
      FC acc;
#pragma unroll
      for (int i = 0; i < 8; ++i)
        acc.e[i] = (float)Gs[(i + 8 * hs_) * 1024 + n0 + nl];
#pragma unroll
      for (int kk = 0; kk < 8; ++kk) {
        const bf16_t* bp = whh + (size_t)n0 * kD + kk * 32;
        __builtin_prefetch(bp + (size_t)16 * kD, 0, 3);  // global_prefetch_b8
        bf16x16 b = frag_b_tr(bp, kD, lane);
        acc.v = wmma_bf16(afr[kk], b, acc.v);
      }
#pragma unroll
      for (int i = 0; i < 8; ++i)
        Gs[(i + 8 * hs_) * 1024 + n0 + nl] = (bf16_t)acc.e[i];
    }
    __syncthreads();

    // ---- phase 2: gates (torch order i,f,g,o) + state update ----
    {
      const int base = tid * 8;   // 16*256 elements / 512 threads
      const int row = base >> 8;
      const int d0 = base & 255;
#pragma unroll
      for (int j = 0; j < 8; ++j) {
        const int d = d0 + j;
        const float gi = (float)Gs[row * 1024 + d];
        const float gf = (float)Gs[row * 1024 + 256 + d];
        const float gg = (float)Gs[row * 1024 + 512 + d];
        const float go = (float)Gs[row * 1024 + 768 + d];
        const float c = sigm(gf) * Cs[row * 256 + d] + sigm(gi) * tanhf(gg);
        const float h = sigm(go) * tanhf(c);
        Cs[row * 256 + d] = c;
        Hs[row * 264 + d] = (bf16_t)h;
        h_out[((size_t)(b0 + row) * kT + t) * kD + d] = h;
      }
    }
    __syncthreads();
  }
}

// ---------------------------------------------------------------------------
// s_norm: batch-axis layernorm of gathered query embeddings. 1 WG, 256 thr.
// ---------------------------------------------------------------------------
__global__ void k_snorm(const int* __restrict__ s, const float* __restrict__ emb,
                        float* __restrict__ snorm) {
  const int d = threadIdx.x;
  float sum = 0.f, ss = 0.f;
  for (int b = 0; b < kB; ++b) {
    const float v = emb[(size_t)s[b] * kD + d];
    sum += v; ss += v * v;
  }
  const float mu = sum * (1.0f / kB);
  const float var = ss * (1.0f / kB) - mu * mu;
  const float inv = rsqrtf(var + 1e-5f);
  for (int b = 0; b < kB; ++b) {
    const float v = emb[(size_t)s[b] * kD + d];
    snorm[b * kD + d] = (v - mu) * inv;
  }
}

// ---------------------------------------------------------------------------
// Fused circular-correlation + Y-projection + attention scores:
//   m = h (*) circulant(s_norm)  (fft-free; B fragment synthesized from LDS)
//   e = tanh(m @ W_y) . w_t      (ds_add_f32 LDS reduction per row)
// grid (32 t-tiles, 64 batches), 512 threads = 16 waves (one n-tile each).
// ---------------------------------------------------------------------------
__global__ void __launch_bounds__(512) k_corr(const float* __restrict__ h,
                                              const float* __restrict__ snorm,
                                              const bf16_t* __restrict__ wyt,
                                              const float* __restrict__ wt,
                                              float* __restrict__ e) {
  __shared__ __align__(16) bf16_t Hs[16 * 264];
  __shared__ __align__(16) bf16_t Mb[16 * 264];
  __shared__ bf16_t Sn[256];
  __shared__ float Ea[16];
  const int tid = threadIdx.x, lane = tid & 31, wave = tid >> 5;
  const int ttile = blockIdx.x, b = blockIdx.y;
  if (tid < 256) Sn[tid] = (bf16_t)snorm[b * kD + tid];
  if (tid < 16) Ea[tid] = 0.0f;
  {   // stage 16 h-rows -> bf16 LDS
    const int m = tid >> 5;
    const int d0 = (tid & 31) * 8;
    const float* src = h + ((size_t)b * kT + ttile * 16 + m) * kD + d0;
    bf16_t* dst = Hs + m * 264 + d0;
#pragma unroll
    for (int j = 0; j < 8; ++j) dst[j] = (bf16_t)src[j];
  }
  __syncthreads();

  const int hs_ = lane >> 4, nl = lane & 15;
  const int n0 = wave * 16;
  {   // phase 1: m-tile
    FC acc;
#pragma unroll
    for (int i = 0; i < 8; ++i) acc.e[i] = 0.0f;
#pragma unroll
    for (int kk = 0; kk < 8; ++kk) {
      bf16x16 a = frag_a_row(Hs + kk * 32, 264, lane);
      bf16x16 bc = frag_b_circ(Sn, n0, kk * 32, lane);
      acc.v = wmma_bf16(a, bc, acc.v);
    }
#pragma unroll
    for (int i = 0; i < 8; ++i)
      Mb[(i + 8 * hs_) * 264 + n0 + nl] = (bf16_t)acc.e[i];
  }
  __syncthreads();
  {   // phase 2: Y = tanh(m @ W_y); e_row += Y * w_t
    FC acc;
#pragma unroll
    for (int i = 0; i < 8; ++i) acc.e[i] = 0.0f;
#pragma unroll
    for (int kk = 0; kk < 8; ++kk) {
      bf16x16 a = frag_a_row(Mb + kk * 32, 264, lane);
      bf16x16 b = frag_b_tr(wyt + (size_t)n0 * kD + kk * 32, kD, lane);
      acc.v = wmma_bf16(a, b, acc.v);
    }
    const float wtn = wt[n0 + nl];
#pragma unroll
    for (int i = 0; i < 8; ++i)
      atomicAdd(&Ea[i + 8 * hs_], tanhf(acc.e[i]) * wtn);   // ds_add_f32
  }
  __syncthreads();
  if (tid < 16) e[(size_t)b * kT + ttile * 16 + tid] = Ea[tid];
}

// ---------------------------------------------------------------------------
// Per-batch softmax over T and r = sum_t a_t * h_t.  grid=64, 256 threads.
// ---------------------------------------------------------------------------
__global__ void k_attend(const float* __restrict__ e, const float* __restrict__ h,
                         float* __restrict__ r) {
  __shared__ float a_sh[512];
  __shared__ float red[256];
  const int tid = threadIdx.x, b = blockIdx.x;
  const float* eb = e + (size_t)b * kT;
  const float v0 = eb[tid], v1 = eb[tid + 256];
  red[tid] = fmaxf(v0, v1);
  __syncthreads();
  for (int s = 128; s > 0; s >>= 1) {
    if (tid < s) red[tid] = fmaxf(red[tid], red[tid + s]);
    __syncthreads();
  }
  const float mx = red[0];
  __syncthreads();
  const float e0 = __expf(v0 - mx), e1 = __expf(v1 - mx);
  red[tid] = e0 + e1;
  __syncthreads();
  for (int s = 128; s > 0; s >>= 1) {
    if (tid < s) red[tid] += red[tid + s];
    __syncthreads();
  }
  const float inv = 1.0f / red[0];
  __syncthreads();
  a_sh[tid] = e0 * inv;
  a_sh[tid + 256] = e1 * inv;
  __syncthreads();
  float racc = 0.f;
  const float* hb = h + (size_t)b * kT * kD;
  for (int t = 0; t < kT; ++t) racc += a_sh[t] * hb[(size_t)t * kD + tid];
  r[b * kD + tid] = racc;
}

// ---------------------------------------------------------------------------
// Final head: rr = tanh(r @ W_p^T + h_last @ W_x^T); y = softmax(rr@W_f^T+b).
// Single WG, 512 threads = 16 waves; 4 m-tiles x 16 n-tiles per GEMM.
// ---------------------------------------------------------------------------
__global__ void __launch_bounds__(512) k_final(const float* __restrict__ r,
                                               const float* __restrict__ h,
                                               const bf16_t* __restrict__ wp,
                                               const bf16_t* __restrict__ wx,
                                               const bf16_t* __restrict__ wf,
                                               const float* __restrict__ bfv,
                                               float* __restrict__ out) {
  __shared__ __align__(16) bf16_t RR[64 * 264];
  const int tid = threadIdx.x, lane = tid & 31, wave = tid >> 5;
  const int hs_ = lane >> 4, nl = lane & 15;

  for (int tix = wave; tix < 64; tix += 16) {   // phase 1: rr
    const int m0 = (tix >> 4) * 16, n0 = (tix & 15) * 16;
    FC acc;
#pragma unroll
    for (int i = 0; i < 8; ++i) acc.e[i] = 0.0f;
#pragma unroll
    for (int kk = 0; kk < 8; ++kk) {
      bf16x16 a1 = frag_a_f32(r + (size_t)m0 * kD + kk * 32, kD, lane);
      bf16x16 b1 = frag_b_tr(wp + (size_t)n0 * kD + kk * 32, kD, lane);
      acc.v = wmma_bf16(a1, b1, acc.v);
      bf16x16 a2 = frag_a_f32(h + ((size_t)m0 * kT + (kT - 1)) * kD + kk * 32,
                              kT * kD, lane);
      bf16x16 b2 = frag_b_tr(wx + (size_t)n0 * kD + kk * 32, kD, lane);
      acc.v = wmma_bf16(a2, b2, acc.v);
    }
#pragma unroll
    for (int i = 0; i < 8; ++i)
      RR[(m0 + i + 8 * hs_) * 264 + n0 + nl] = (bf16_t)tanhf(acc.e[i]);
  }
  __syncthreads();

  for (int tix = wave; tix < 64; tix += 16) {   // phase 2: logits -> out
    const int m0 = (tix >> 4) * 16, n0 = (tix & 15) * 16;
    FC acc;
    const float bias = bfv[n0 + nl];
#pragma unroll
    for (int i = 0; i < 8; ++i) acc.e[i] = bias;
#pragma unroll
    for (int kk = 0; kk < 8; ++kk) {
      bf16x16 a = frag_a_row(RR + m0 * 264 + kk * 32, 264, lane);
      bf16x16 b = frag_b_tr(wf + (size_t)n0 * kD + kk * 32, kD, lane);
      acc.v = wmma_bf16(a, b, acc.v);
    }
#pragma unroll
    for (int i = 0; i < 8; ++i)
      out[(size_t)(m0 + i + 8 * hs_) * kD + n0 + nl] = acc.e[i];
  }
  __threadfence();
  __syncthreads();

  if (tid < 64) {   // phase 3: row softmax in place
    float* row = out + (size_t)tid * kD;
    float mx = -3.4e38f;
    for (int d = 0; d < kD; ++d) mx = fmaxf(mx, row[d]);
    float s = 0.f;
    for (int d = 0; d < kD; ++d) s += __expf(row[d] - mx);
    const float inv = 1.0f / s;
    for (int d = 0; d < kD; ++d) row[d] = __expf(row[d] - mx) * inv;
  }
}

// ---------------------------------------------------------------------------
// Host launcher
// ---------------------------------------------------------------------------
extern "C" void kernel_launch(void* const* d_in, const int* in_sizes, int n_in,
                              void* d_out, int out_size, void* d_ws, size_t ws_size,
                              hipStream_t stream) {
  (void)in_sizes; (void)n_in; (void)out_size; (void)ws_size;
  const int*   x    = (const int*)d_in[0];
  const int*   s    = (const int*)d_in[1];
  const float* emb  = (const float*)d_in[2];
  const float* w_ih = (const float*)d_in[3];
  const float* w_hh = (const float*)d_in[4];
  const float* b_l  = (const float*)d_in[5];
  const float* W_y  = (const float*)d_in[6];
  const float* w_t  = (const float*)d_in[7];
  const float* W_p  = (const float*)d_in[8];
  const float* W_x  = (const float*)d_in[9];
  const float* W_f  = (const float*)d_in[10];
  const float* b_f  = (const float*)d_in[11];

  char* ws = (char*)d_ws;
  bf16_t* wih_b = (bf16_t*)(ws + OFF_WIH);
  bf16_t* whh_b = (bf16_t*)(ws + OFF_WHH);
  bf16_t* wyt_b = (bf16_t*)(ws + OFF_WYT);
  bf16_t* wp_b  = (bf16_t*)(ws + OFF_WP);
  bf16_t* wx_b  = (bf16_t*)(ws + OFF_WX);
  bf16_t* wf_b  = (bf16_t*)(ws + OFF_WF);
  bf16_t* xs_b  = (bf16_t*)(ws + OFF_XS);
  float*  h_f   = (float*)(ws + OFF_H);
  float*  sn_f  = (float*)(ws + OFF_SN);
  float*  e_f   = (float*)(ws + OFF_E);
  float*  r_f   = (float*)(ws + OFF_R);

  // weight conversions (bf16; w_ih/w_hh/W_p/W_x/W_f rows are already Wt[n][k])
  k_cvt  <<<1024, 256, 0, stream>>>(w_ih, wih_b, kG * kD);
  k_cvt  <<<1024, 256, 0, stream>>>(w_hh, whh_b, kG * kD);
  k_cvt_t<<< 256, 256, 0, stream>>>(W_y, wyt_b);
  k_cvt  <<< 256, 256, 0, stream>>>(W_p, wp_b, kD * kD);
  k_cvt  <<< 256, 256, 0, stream>>>(W_x, wx_b, kD * kD);
  k_cvt  <<< 256, 256, 0, stream>>>(W_f, wf_b, kD * kD);

  // embedding gather + input projection (17.2 GFLOP WMMA GEMM)
  k_xproj<<<dim3(kB * kT / 16, 8), 256, 0, stream>>>(x, emb, wih_b, b_l, xs_b);

  // persistent LSTM scan (LDS-resident state, async xs DMA, 4 batch shards)
  k_lstm<<<4, 512, 0, stream>>>(xs_b, whh_b, h_f);

  // query normalization (batch-axis)
  k_snorm<<<1, 256, 0, stream>>>(s, emb, sn_f);

  // fused circular correlation + Y projection + attention scores
  k_corr<<<dim3(kT / 16, kB), 512, 0, stream>>>(h_f, sn_f, wyt_b, w_t, e_f);

  // softmax over T + attention-weighted sum
  k_attend<<<kB, 256, 0, stream>>>(e_f, h_f, r_f);

  // final head + softmax into d_out
  k_final<<<1, 512, 0, stream>>>(r_f, h_f, wp_b, wx_b, wf_b, b_f, (float*)d_out);
}
